// custom_rnn_model_74612171866344
// MI455X (gfx1250) — matrix operand order repaired and checked by reading, not compiled, not run on hardware
//
#include <hip/hip_runtime.h>
#include <hip/hip_bf16.h>

typedef __attribute__((ext_vector_type(16))) _Float16 v16h;
typedef __attribute__((ext_vector_type(8)))  _Float16 v8h;
typedef __attribute__((ext_vector_type(4)))  _Float16 v4h;
typedef __attribute__((ext_vector_type(8)))  float    v8f;
typedef __attribute__((ext_vector_type(4)))  float    v4f;

#define SAMPLES 100000
#define DIN     512
#define DHID    512
#define DOUT    128
#define TILE_M  128   // rows per workgroup (8 waves x 16-row M-tiles)

// ---------------------------------------------------------------------------
// Prepass: convert fp32 weights [K x N] (row-major, row = K) into f16 WMMA
// B-fragments. Tile (kt, nt) covers K = kt*32..+31, N = nt*16..+15.
// Per-lane layout (v_wmma_f32_16x16x32_f16 B operand, wave32):
//   lane l: column N = nt*16 + (l & 15)
//           K values: kt*32 + (l < 16 ? 0 : 8) + (h < 8 ? h : h + 8), h = 0..15, packed
//           2-per-VGPR -> 16 contiguous halfs per lane.
// Fragment storage: dst[tile*512 + lane*16 + h]
// ---------------------------------------------------------------------------
__global__ void pack_weights_f16(const float* __restrict__ W,
                                 _Float16* __restrict__ dst, int K, int N) {
  int t = blockIdx.x * blockDim.x + threadIdx.x;
  int ntiles = N >> 4;
  int tiles  = (K >> 5) * ntiles;
  if (t >= tiles * 32) return;
  int tile = t >> 5;
  int lane = t & 31;
  int kt = tile / ntiles;
  int nt = tile % ntiles;
  int col  = (nt << 4) + (lane & 15);
  int krow = (kt << 5) + ((lane >> 4) << 3);
  _Float16* o = dst + (size_t)tile * 512 + (size_t)lane * 16;
#pragma unroll
  for (int h = 0; h < 16; ++h)
    o[h] = (_Float16)W[(size_t)(krow + (h < 8 ? h : h + 8)) * N + col];
}

// ---------------------------------------------------------------------------
// A-fragment load from a row-major f16 LDS tile [TILE_M x 512].
// 16-bit A 16x32 layout: lane<16 holds row M=lane, K = k0+{0..7, 16..23};
// lane>=16 holds row M=lane-16, K = k0+{8..15, 24..31}.
// ---------------------------------------------------------------------------
static __device__ __forceinline__ v16h load_afrag(const _Float16* __restrict__ buf,
                                                  int mt, int kt, int lane) {
  int row  = mt * 16 + (lane & 15);
  int koff = kt * 32 + ((lane >> 4) << 3);
  const _Float16* p = buf + row * 512 + koff;
  v8h lo = *(const v8h*)p;          // K = koff .. koff+7
  v8h hi = *(const v8h*)(p + 16);   // K = koff+16 .. koff+23
  return __builtin_shufflevector(lo, hi, 0, 1, 2, 3, 4, 5, 6, 7,
                                         8, 9, 10, 11, 12, 13, 14, 15);
}

static __device__ __forceinline__ v8f wmma_f16(v16h a, v16h b, v8f c) {
  return __builtin_amdgcn_wmma_f32_16x16x32_f16(false, a, false, b,
                                                (short)0, c, false, false);
}

// ---------------------------------------------------------------------------
// One 128x512 = [128x512] x [512x512] GEMM stage, LDS -> LDS.
// Wave w owns N-tiles {2w, 2w+1} and {16+2w, 17+2w}; all 8 M-tiles per wave,
// so each B fragment loaded from L2 feeds 8 WMMAs.
// ---------------------------------------------------------------------------
template <bool RELU>
static __device__ __forceinline__ void gemm512(const _Float16* __restrict__ src,
                                               _Float16* __restrict__ dst,
                                               const _Float16* __restrict__ Wf,
                                               const float* __restrict__ bias,
                                               int wave, int lane) {
  const int col0 = lane & 15;
  const int sel  = lane >> 4;
  const v8f zero = {0.f, 0.f, 0.f, 0.f, 0.f, 0.f, 0.f, 0.f};

#pragma unroll 1
  for (int chunk = 0; chunk < 2; ++chunk) {
    const int nt0 = wave * 2 + chunk * 16;
    v8f c[8][2];
#pragma unroll
    for (int m = 0; m < 8; ++m) { c[m][0] = zero; c[m][1] = zero; }

#pragma unroll 1
    for (int kt = 0; kt < 16; ++kt) {
      // tile linear index = kt*32 + nt ; frag = tile*512 + lane*16
      v16h b0 = *(const v16h*)(Wf + ((size_t)(kt * 32 + nt0)     * 32 + lane) * 16);
      v16h b1 = *(const v16h*)(Wf + ((size_t)(kt * 32 + nt0 + 1) * 32 + lane) * 16);
#pragma unroll
      for (int m = 0; m < 8; ++m) {
        v16h a = load_afrag(src, m, kt, lane);
        c[m][0] = wmma_f16(a, b0, c[m][0]);
        c[m][1] = wmma_f16(a, b1, c[m][1]);
      }
    }

    // Epilogue: (optional bias+ReLU), convert to f16, store to LDS row-major.
#pragma unroll
    for (int n = 0; n < 2; ++n) {
      const int col = (nt0 + n) * 16 + col0;
      float bv = 0.f;
      if constexpr (RELU) bv = bias[col];
#pragma unroll
      for (int m = 0; m < 8; ++m) {
#pragma unroll
        for (int r = 0; r < 8; ++r) {
          float v = c[m][n][r];
          if constexpr (RELU) v = fmaxf(v + bv, 0.f);
          dst[(m * 16 + sel * 8 + r) * 512 + col] = (_Float16)v;
        }
      }
    }
  }
}

// ---------------------------------------------------------------------------
// Fused chain: X(+b_in) -> f16 LDS -> GEMM1 -> GEMM2(+b_rec, relu) -> GEMM3
// (+b_out) -> global f32. Intermediates never touch HBM.
// ---------------------------------------------------------------------------
__global__ __launch_bounds__(256, 1) void rnn_fused(
    const float* __restrict__ X, const float* __restrict__ b_in,
    const float* __restrict__ b_rec, const float* __restrict__ b_out,
    const _Float16* __restrict__ Wf_in, const _Float16* __restrict__ Wf_rec,
    const _Float16* __restrict__ Wf_out, float* __restrict__ Y) {
  extern __shared__ _Float16 smem[];
  _Float16* bufA = smem;                 // 128 x 512 f16 = 128 KB
  _Float16* bufB = smem + TILE_M * 512;  // 128 x 512 f16 = 128 KB

  const int tid  = threadIdx.x;
  const int wave = tid >> 5;
  const int lane = tid & 31;
  const int base_row = blockIdx.x * TILE_M;

  // Stage 0: global fp32 X tile -> (+b_in) -> f16 LDS, coalesced float4 loads.
  for (int i = 0; i < (TILE_M * 512 / 4) / 256; ++i) {
    int idx = i * 256 + tid;
    int row = idx >> 7;              // 128 float4 groups per row
    int k4  = (idx & 127) << 2;
    int rg  = base_row + row;
    if (rg > SAMPLES - 1) rg = SAMPLES - 1;  // clamp tail (rows independent)
    v4f x  = *(const v4f*)(X + (size_t)rg * DIN + k4);
    v4f bb = *(const v4f*)(b_in + k4);
    v4h h;
    h[0] = (_Float16)(x[0] + bb[0]);
    h[1] = (_Float16)(x[1] + bb[1]);
    h[2] = (_Float16)(x[2] + bb[2]);
    h[3] = (_Float16)(x[3] + bb[3]);
    *(v4h*)(bufA + row * 512 + k4) = h;
  }
  __syncthreads();

  gemm512<false>(bufA, bufB, Wf_in, nullptr, wave, lane);   // inp
  __syncthreads();
  gemm512<true>(bufB, bufA, Wf_rec, b_rec, wave, lane);     // hidden = relu(..)
  __syncthreads();

  // GEMM3: [128x512] x [512x128] -> global. Wave w owns N-tile w (8 tiles).
  {
    const int col0 = lane & 15;
    const int sel  = lane >> 4;
    const v8f zero = {0.f, 0.f, 0.f, 0.f, 0.f, 0.f, 0.f, 0.f};
    v8f c[8];
#pragma unroll
    for (int m = 0; m < 8; ++m) c[m] = zero;

#pragma unroll 1
    for (int kt = 0; kt < 16; ++kt) {
      v16h b = *(const v16h*)(Wf_out + ((size_t)(kt * 8 + wave) * 32 + lane) * 16);
#pragma unroll
      for (int m = 0; m < 8; ++m) {
        v16h a = load_afrag(bufA, m, kt, lane);
        c[m] = wmma_f16(a, b, c[m]);
      }
    }

    const int col = wave * 16 + col0;
    const float bv = b_out[col];
#pragma unroll
    for (int m = 0; m < 8; ++m) {
#pragma unroll
      for (int r = 0; r < 8; ++r) {
        int rg = base_row + m * 16 + sel * 8 + r;
        if (rg < SAMPLES) Y[(size_t)rg * DOUT + col] = c[m][r] + bv;
      }
    }
  }
}

// ---------------------------------------------------------------------------
extern "C" void kernel_launch(void* const* d_in, const int* in_sizes, int n_in,
                              void* d_out, int out_size, void* d_ws, size_t ws_size,
                              hipStream_t stream) {
  (void)in_sizes; (void)n_in; (void)out_size; (void)ws_size;
  const float* X    = (const float*)d_in[0];
  const float* Win  = (const float*)d_in[1];
  const float* bin  = (const float*)d_in[2];
  const float* Wrec = (const float*)d_in[3];
  const float* brec = (const float*)d_in[4];
  const float* Wout = (const float*)d_in[5];
  const float* bout = (const float*)d_in[6];
  float* Y = (float*)d_out;

  // Workspace: f16 fragment-packed weights (512KB + 512KB + 128KB).
  _Float16* wf_in  = (_Float16*)d_ws;
  _Float16* wf_rec = wf_in + 512 * 512;
  _Float16* wf_out = wf_rec + 512 * 512;

  pack_weights_f16<<<64, 256, 0, stream>>>(Win,  wf_in,  512, 512);
  pack_weights_f16<<<64, 256, 0, stream>>>(Wrec, wf_rec, 512, 512);
  pack_weights_f16<<<16, 256, 0, stream>>>(Wout, wf_out, 512, 128);

  const int grid = (SAMPLES + TILE_M - 1) / TILE_M;           // 782
  const size_t lds = 2 * (size_t)TILE_M * 512 * sizeof(_Float16);  // 256 KB
  rnn_fused<<<grid, 256, lds, stream>>>(X, bin, brec, bout,
                                        wf_in, wf_rec, wf_out, Y);
}
